// MultiHeadAttentionBlock_60327110639992
// MI455X (gfx1250) — matrix operand order repaired and checked by reading, not compiled, not run on hardware
//
#include <hip/hip_runtime.h>
#include <hip/hip_bf16.h>

#define BB 4
#define SS 2048
#define DD 1024
#define HH 16
#define HDIM 64
#define LK 72   // padded LDS row stride (halfwords); 144B, multiple of 16B

typedef __attribute__((ext_vector_type(16))) __bf16 v16bf;
typedef __attribute__((ext_vector_type(8)))  float  v8f;

union Frag16 { v16bf v; uint4 u[2]; __bf16 h[16]; };

static __device__ inline unsigned pack_bf2(float a, float b) {
  unsigned short x = __builtin_bit_cast(unsigned short, (__bf16)a);
  unsigned short y = __builtin_bit_cast(unsigned short, (__bf16)b);
  return (unsigned)x | ((unsigned)y << 16);
}

// CDNA5 async global->LDS copy (ASYNCcnt-tracked, bypasses VGPRs).
// addrspace(3) flat addresses keep the LDS byte offset in the low 32 bits
// (ISA 10.2 aperture rules), so truncating the shared pointer is the VDST offset.
static __device__ inline void async_copy_b128(unsigned lds_off, const void* gaddr) {
  asm volatile("global_load_async_to_lds_b128 %0, %1, off"
               :: "v"(lds_off), "v"(gaddr)
               : "memory");
}
static __device__ inline void wait_async0() {
  asm volatile("s_wait_asynccnt 0x0" ::: "memory");
}

// ---------------------------------------------------------------------------
// Projection: out = bf16( x @ W^T + bias ).  VTRANS==false -> [b][h][s][hd]
// (Q,K); VTRANS==true -> [b][h][hd][s] (V, pre-transposed for the PV matmul,
// which also makes the epilogue stores contiguous b128 along s).
// 128 threads (4 waves), 64x64 tile, K stepped by 32, wave = 2x2 fragments.
// ---------------------------------------------------------------------------
template <bool VTRANS>
__global__ __launch_bounds__(128) void proj_kernel(
    const float* __restrict__ x, const float* __restrict__ w,
    const float* __restrict__ bias, __bf16* __restrict__ out)
{
  __shared__ __align__(16) __bf16 lds_a[64 * 40];  // [m][k] stride 40 (80B)
  __shared__ __align__(16) __bf16 lds_b[64 * 40];  // [e][k] stride 40

  const int tid  = threadIdx.x;
  const int wv   = tid >> 5;
  const int lane = tid & 31;
  const int li   = lane & 15;
  const int g    = lane >> 4;
  const int wm   = wv >> 1, wn = wv & 1;
  const int tileM = blockIdx.y * 64;
  const int tileN = blockIdx.x * 64;

  v8f acc[2][2] = {};

  const int ldr  = tid >> 1;  // 0..63 row within tile
  const int half = tid & 1;   // 16-col group

  for (int kk = 0; kk < DD; kk += 32) {
    {
      const float* src = x + (size_t)(tileM + ldr) * DD + kk + half * 16;
      float4 f0 = ((const float4*)src)[0];
      float4 f1 = ((const float4*)src)[1];
      float4 f2 = ((const float4*)src)[2];
      float4 f3 = ((const float4*)src)[3];
      uint4 p0, p1;
      p0.x = pack_bf2(f0.x, f0.y); p0.y = pack_bf2(f0.z, f0.w);
      p0.z = pack_bf2(f1.x, f1.y); p0.w = pack_bf2(f1.z, f1.w);
      p1.x = pack_bf2(f2.x, f2.y); p1.y = pack_bf2(f2.z, f2.w);
      p1.z = pack_bf2(f3.x, f3.y); p1.w = pack_bf2(f3.z, f3.w);
      *(uint4*)(lds_a + ldr * 40 + half * 16)     = p0;
      *(uint4*)(lds_a + ldr * 40 + half * 16 + 8) = p1;
    }
    {
      const float* src = w + (size_t)(tileN + ldr) * DD + kk + half * 16;
      float4 f0 = ((const float4*)src)[0];
      float4 f1 = ((const float4*)src)[1];
      float4 f2 = ((const float4*)src)[2];
      float4 f3 = ((const float4*)src)[3];
      uint4 p0, p1;
      p0.x = pack_bf2(f0.x, f0.y); p0.y = pack_bf2(f0.z, f0.w);
      p0.z = pack_bf2(f1.x, f1.y); p0.w = pack_bf2(f1.z, f1.w);
      p1.x = pack_bf2(f2.x, f2.y); p1.y = pack_bf2(f2.z, f2.w);
      p1.z = pack_bf2(f3.x, f3.y); p1.w = pack_bf2(f3.z, f3.w);
      *(uint4*)(lds_b + ldr * 40 + half * 16)     = p0;
      *(uint4*)(lds_b + ldr * 40 + half * 16 + 8) = p1;
    }
    __syncthreads();

    #pragma unroll
    for (int mi = 0; mi < 2; ++mi) {
      const __bf16* ar = lds_a + (wm * 32 + mi * 16 + li) * 40;
      Frag16 af;
      af.u[0] = *(const uint4*)(ar + 8 * g);
      af.u[1] = *(const uint4*)(ar + 16 + 8 * g);
      #pragma unroll
      for (int ni = 0; ni < 2; ++ni) {
        const __bf16* br = lds_b + (wn * 32 + ni * 16 + li) * 40 + 8 * g;
        Frag16 bf;
        bf.u[0] = *(const uint4*)(br);
        bf.u[1] = *(const uint4*)(br + 16);
        acc[mi][ni] = __builtin_amdgcn_wmma_f32_16x16x32_bf16(
            false, af.v, false, bf.v, (short)0, acc[mi][ni], false, false);
      }
    }
    __syncthreads();
  }

  // epilogue: +bias, branch-free layout chosen at compile time
  #pragma unroll
  for (int mi = 0; mi < 2; ++mi) {
    const int rowg0 = tileM + wm * 32 + mi * 16 + 8 * g;   // row of element r=0
    const int bb = rowg0 / SS, ss0 = rowg0 % SS;           // 64-tiles never cross b
    #pragma unroll
    for (int ni = 0; ni < 2; ++ni) {
      const int e  = tileN + wn * 32 + ni * 16 + li;
      const float bv_ = bias[e];
      const int hh = e / HDIM, hd = e % HDIM;
      if (VTRANS) {
        // rows r=0..7 are 8 consecutive s -> single b128 store
        __bf16* dst = out + (((size_t)bb * HH + hh) * HDIM + hd) * SS + ss0;
        uint4 pk;
        pk.x = pack_bf2(acc[mi][ni][0] + bv_, acc[mi][ni][1] + bv_);
        pk.y = pack_bf2(acc[mi][ni][2] + bv_, acc[mi][ni][3] + bv_);
        pk.z = pack_bf2(acc[mi][ni][4] + bv_, acc[mi][ni][5] + bv_);
        pk.w = pack_bf2(acc[mi][ni][6] + bv_, acc[mi][ni][7] + bv_);
        *(uint4*)dst = pk;
      } else {
        __bf16* dst = out + (((size_t)bb * HH + hh) * SS + ss0) * HDIM + hd;
        #pragma unroll
        for (int r = 0; r < 8; ++r)
          dst[(size_t)r * HDIM] = (__bf16)(acc[mi][ni][r] + bv_);
      }
    }
  }
}

// ---------------------------------------------------------------------------
// Flash attention: block = (b, h, 64 q rows); 4 waves x 16 q rows.
// K/V tiles double-buffered in LDS via async global->LDS b128 copies.
// Vt is stored [b][h][hd][s] so the V tile lands directly in [hd][kpos].
// ---------------------------------------------------------------------------
__global__ __launch_bounds__(128) void attn_kernel(
    const __bf16* __restrict__ Qh, const __bf16* __restrict__ Kh,
    const __bf16* __restrict__ Vt, const float* __restrict__ maskp,
    float* __restrict__ out)
{
  __shared__ __align__(16) __bf16 lds_k[2][64 * LK];   // [kpos][hd]
  __shared__ __align__(16) __bf16 lds_v[2][64 * LK];   // [hd][kpos]
  __shared__ __align__(16) __bf16 lds_p[4][16 * LK];   // per-wave P tile

  const int tid  = threadIdx.x;
  const int wv   = tid >> 5;
  const int lane = tid & 31;
  const int li   = lane & 15;
  const int g    = lane >> 4;
  const int b = blockIdx.z, h = blockIdx.y, qt = blockIdx.x;
  const size_t off = ((size_t)(b * HH + h)) * SS * HDIM;
  const int q0 = qt * 64 + wv * 16;

  // Q fragments (A layout), resident for the whole k loop
  Frag16 qa[2];
  {
    const __bf16* qr = Qh + off + (size_t)(q0 + li) * HDIM;
    #pragma unroll
    for (int s = 0; s < 2; ++s) {
      qa[s].u[0] = *(const uint4*)(qr + 32 * s + 8 * g);
      qa[s].u[1] = *(const uint4*)(qr + 32 * s + 16 + 8 * g);
    }
  }

  // async issue of one 64x64 K tile and one 64x64 V tile (8 b128 per thread)
  auto issue_tile = [&](int kt, int buf) {
    #pragma unroll
    for (int j = 0; j < 4; ++j) {
      const int idx = tid * 4 + j;            // 0..511
      const int r = idx >> 3, qu = idx & 7;   // row, 8-halfword chunk
      const __bf16* gk = Kh + off + (size_t)(kt * 64 + r) * HDIM + qu * 8;
      async_copy_b128((unsigned)(size_t)(const void*)&lds_k[buf][r * LK + qu * 8], gk);
      const __bf16* gv = Vt + off + (size_t)r * SS + kt * 64 + qu * 8;
      async_copy_b128((unsigned)(size_t)(const void*)&lds_v[buf][r * LK + qu * 8], gv);
    }
  };

  v8f acc[4] = {};
  float rowmax[8], rowsum[8];
  #pragma unroll
  for (int r = 0; r < 8; ++r) { rowmax[r] = -3.0e38f; rowsum[r] = 0.0f; }

  issue_tile(0, 0);

  for (int kt = 0; kt < SS / 64; ++kt) {
    const int buf = kt & 1;
    wait_async0();        // this wave's tile landed in LDS
    __syncthreads();      // all waves' tiles landed; prev reads of buf^1 done
    if (kt + 1 < SS / 64) issue_tile(kt + 1, buf ^ 1);

    // scores = Q @ K^T (4 n-frags x 2 k-steps)
    v8f sf[4] = {};
    #pragma unroll
    for (int nt = 0; nt < 4; ++nt) {
      #pragma unroll
      for (int s = 0; s < 2; ++s) {
        const __bf16* kr = &lds_k[buf][(nt * 16 + li) * LK + 32 * s + 8 * g];
        Frag16 bf;
        bf.u[0] = *(const uint4*)(kr);
        bf.u[1] = *(const uint4*)(kr + 16);
        sf[nt] = __builtin_amdgcn_wmma_f32_16x16x32_bf16(
            false, qa[s].v, false, bf.v, (short)0, sf[nt], false, false);
      }
      const int kcol = kt * 64 + nt * 16 + li;
      const float madd = (1.0f - maskp[b * SS + kcol]) * -1.0e30f;
      #pragma unroll
      for (int r = 0; r < 8; ++r)
        sf[nt][r] = sf[nt][r] * 0.125f + madd;   // 1/sqrt(64)
    }

    // online softmax; row r+8g lives in 16 lanes -> xor-reduce 1/2/4/8
    #pragma unroll
    for (int r = 0; r < 8; ++r) {
      float t = fmaxf(fmaxf(sf[0][r], sf[1][r]), fmaxf(sf[2][r], sf[3][r]));
      t = fmaxf(t, __shfl_xor(t, 1, 32));
      t = fmaxf(t, __shfl_xor(t, 2, 32));
      t = fmaxf(t, __shfl_xor(t, 4, 32));
      t = fmaxf(t, __shfl_xor(t, 8, 32));
      const float nmax  = fmaxf(rowmax[r], t);
      const float alpha = __expf(rowmax[r] - nmax);
      rowmax[r] = nmax;
      float ps = 0.0f;
      #pragma unroll
      for (int nt = 0; nt < 4; ++nt) {
        const float p = __expf(sf[nt][r] - nmax);
        sf[nt][r] = p;
        ps += p;
      }
      ps += __shfl_xor(ps, 1, 32);
      ps += __shfl_xor(ps, 2, 32);
      ps += __shfl_xor(ps, 4, 32);
      ps += __shfl_xor(ps, 8, 32);
      rowsum[r] = rowsum[r] * alpha + ps;
      #pragma unroll
      for (int f = 0; f < 4; ++f) acc[f][r] *= alpha;
    }

    // P (D layout) -> LDS bf16 -> reload in A layout
    #pragma unroll
    for (int nt = 0; nt < 4; ++nt)
      #pragma unroll
      for (int r = 0; r < 8; ++r)
        lds_p[wv][(r + 8 * g) * LK + nt * 16 + li] = (__bf16)sf[nt][r];
    __syncthreads();

    Frag16 pa[2];
    #pragma unroll
    for (int ks = 0; ks < 2; ++ks) {
      const __bf16* pr = lds_p[wv] + li * LK + 32 * ks;
      pa[ks].u[0] = *(const uint4*)(pr + 8 * g);
      pa[ks].u[1] = *(const uint4*)(pr + 16 + 8 * g);
    }
    // ctx += P @ V (4 hd-frags x 2 k-steps)
    #pragma unroll
    for (int ht = 0; ht < 4; ++ht) {
      #pragma unroll
      for (int ks = 0; ks < 2; ++ks) {
        const __bf16* vr = &lds_v[buf][(ht * 16 + li) * LK + 32 * ks + 8 * g];
        Frag16 bf;
        bf.u[0] = *(const uint4*)(vr);
        bf.u[1] = *(const uint4*)(vr + 16);
        acc[ht] = __builtin_amdgcn_wmma_f32_16x16x32_bf16(
            false, pa[ks].v, false, bf.v, (short)0, acc[ht], false, false);
      }
    }
  }

  // normalize and write ctx as (B,S,D) f32
  #pragma unroll
  for (int ht = 0; ht < 4; ++ht) {
    const int hd = ht * 16 + li;
    float* dst = out + ((size_t)b * SS + q0 + 8 * g) * DD + h * HDIM + hd;
    #pragma unroll
    for (int r = 0; r < 8; ++r)
      dst[(size_t)r * DD] = acc[ht][r] / rowsum[r];
  }
}

extern "C" void kernel_launch(void* const* d_in, const int* in_sizes, int n_in,
                              void* d_out, int out_size, void* d_ws, size_t ws_size,
                              hipStream_t stream) {
  (void)in_sizes; (void)n_in; (void)out_size; (void)ws_size;
  const float* q    = (const float*)d_in[0];
  const float* k    = (const float*)d_in[1];
  const float* v    = (const float*)d_in[2];
  const float* mask = (const float*)d_in[3];
  const float* wq   = (const float*)d_in[4];
  const float* bq   = (const float*)d_in[5];
  const float* wk   = (const float*)d_in[6];
  const float* bk   = (const float*)d_in[7];
  const float* wv   = (const float*)d_in[8];
  const float* bv   = (const float*)d_in[9];

  const size_t N = (size_t)BB * HH * SS * HDIM;
  __bf16* Qh = (__bf16*)d_ws;
  __bf16* Kh = Qh + N;
  __bf16* Vt = Kh + N;   // [b][h][hd][s]

  dim3 pb(128), pg(DD / 64, (BB * SS) / 64);
  proj_kernel<false><<<pg, pb, 0, stream>>>(q, wq, bq, Qh);
  proj_kernel<false><<<pg, pb, 0, stream>>>(k, wk, bk, Kh);
  proj_kernel<true ><<<pg, pb, 0, stream>>>(v, wv, bv, Vt);

  dim3 ab(128), ag(SS / 64, HH, BB);
  attn_kernel<<<ag, ab, 0, stream>>>(Qh, Kh, Vt, mask, (float*)d_out);
}
